// SelfAttention_1769526526119
// MI455X (gfx1250) — matrix operand order repaired and checked by reading, not compiled, not run on hardware
//
#include <hip/hip_runtime.h>
#include <hip/hip_bf16.h>

typedef __attribute__((ext_vector_type(16))) __bf16 v16bf;
typedef __attribute__((ext_vector_type(8)))  float  v8f;

union BF16Frag {
    v16bf v;
    uint4 q[2];
    unsigned short u[16];
};

#define NTOK 4096
#define CIN  256
#define MID  32
#define LOG2E 1.44269504f

static __device__ __forceinline__ unsigned short f32_bf16(float f) {
    union { float f; unsigned int u; } v; v.f = f;
    unsigned int r = v.u + 0x7FFFu + ((v.u >> 16) & 1u);  // round-nearest-even
    return (unsigned short)(r >> 16);
}
static __device__ __forceinline__ unsigned int pack_bf16x2(float lo, float hi) {
    return (unsigned int)f32_bf16(lo) | ((unsigned int)f32_bf16(hi) << 16);
}

// ---- fused DPP16 butterfly reductions: one VOP2+DPP instruction per stage --
static __device__ __forceinline__ float redmax16(float x) {
    float t;
    asm("v_max_num_f32_dpp %0, %1, %1 quad_perm:[1,0,3,2] row_mask:0xf bank_mask:0xf bound_ctrl:1"
        : "=v"(t) : "v"(x));
    asm("v_max_num_f32_dpp %0, %1, %1 quad_perm:[2,3,0,1] row_mask:0xf bank_mask:0xf bound_ctrl:1"
        : "=v"(x) : "v"(t));
    asm("v_max_num_f32_dpp %0, %1, %1 row_half_mirror row_mask:0xf bank_mask:0xf bound_ctrl:1"
        : "=v"(t) : "v"(x));
    asm("v_max_num_f32_dpp %0, %1, %1 row_mirror row_mask:0xf bank_mask:0xf bound_ctrl:1"
        : "=v"(x) : "v"(t));
    return x;
}
static __device__ __forceinline__ float redsum16(float x) {
    float t;
    asm("v_add_f32_dpp %0, %1, %1 quad_perm:[1,0,3,2] row_mask:0xf bank_mask:0xf bound_ctrl:1"
        : "=v"(t) : "v"(x));
    asm("v_add_f32_dpp %0, %1, %1 quad_perm:[2,3,0,1] row_mask:0xf bank_mask:0xf bound_ctrl:1"
        : "=v"(x) : "v"(t));
    asm("v_add_f32_dpp %0, %1, %1 row_half_mirror row_mask:0xf bank_mask:0xf bound_ctrl:1"
        : "=v"(t) : "v"(x));
    asm("v_add_f32_dpp %0, %1, %1 row_mirror row_mask:0xf bank_mask:0xf bound_ctrl:1"
        : "=v"(x) : "v"(t));
    return x;
}

// ---------------------------------------------------------------------------
// Kernel 1: 1x1-conv projections f,g,h with BN+ReLU (f,g). Outputs bf16.
//   fT : [b][n][m]       (A-fragment layout for S = f^T g, includes mid^-0.5)
//   gT : [b][n][m]       (B-fragment layout: column p contiguous over m)
//   hM : [b][m][perm(n)] (B-fragment layout with per-32-block K permutation:
//                         phys 2t <- logical t, phys 2t+1 <- logical t+16,
//                         matching the packed-P LDS layout in attn_kernel)
// ---------------------------------------------------------------------------
__global__ __launch_bounds__(256) void proj_kernel(
    const float* __restrict__ x,
    const float* __restrict__ f_w, const float* __restrict__ f_b,
    const float* __restrict__ f_gamma, const float* __restrict__ f_beta,
    const float* __restrict__ f_mean, const float* __restrict__ f_var,
    const float* __restrict__ g_w, const float* __restrict__ g_b,
    const float* __restrict__ g_gamma, const float* __restrict__ g_beta,
    const float* __restrict__ g_mean, const float* __restrict__ g_var,
    const float* __restrict__ h_w, const float* __restrict__ h_b,
    unsigned short* __restrict__ fT, unsigned short* __restrict__ gT,
    unsigned short* __restrict__ hM)
{
    int t = blockIdx.x * blockDim.x + threadIdx.x;   // one thread per (b,n)
    int b = t >> 12;
    int n = t & (NTOK - 1);
    const float* xb = x + ((size_t)b * CIN) * NTOK + n;

    // K-permuted position of n inside its 32-block (see header comment)
    int t32 = n & 31;
    int nperm = (n & ~31) | ((t32 < 16) ? (2 * t32) : (2 * (t32 - 16) + 1));

    for (int mg = 0; mg < MID / 8; ++mg) {
        float af[8] = {0}, ag[8] = {0}, ah[8] = {0};
        for (int c = 0; c < CIN; ++c) {
            float xv = xb[(size_t)c * NTOK];
#pragma unroll
            for (int mi = 0; mi < 8; ++mi) {
                int m = mg * 8 + mi;
                af[mi] = fmaf(f_w[m * CIN + c], xv, af[mi]);
                ag[mi] = fmaf(g_w[m * CIN + c], xv, ag[mi]);
                ah[mi] = fmaf(h_w[m * CIN + c], xv, ah[mi]);
            }
        }
#pragma unroll
        for (int mi = 0; mi < 8; ++mi) {
            int m = mg * 8 + mi;
            float rf = rsqrtf(f_var[m] + 1e-5f);
            float fv = (af[mi] + f_b[m]) * (f_gamma[m] * rf) +
                       (f_beta[m] - f_gamma[m] * f_mean[m] * rf);
            fv = fmaxf(fv, 0.0f) * 0.17677669529663688f;   // * mid^-0.5
            float rg = rsqrtf(g_var[m] + 1e-5f);
            float gv = (ag[mi] + g_b[m]) * (g_gamma[m] * rg) +
                       (g_beta[m] - g_gamma[m] * g_mean[m] * rg);
            gv = fmaxf(gv, 0.0f);
            float hv = ah[mi] + h_b[m];
            fT[(size_t)t * MID + m] = f32_bf16(fv);
            gT[(size_t)t * MID + m] = f32_bf16(gv);
            hM[((size_t)(b * MID + m)) * NTOK + nperm] = f32_bf16(hv);
        }
    }
}

// ---------------------------------------------------------------------------
// Kernel 2: flash-attention. One wave = 16 q-rows; p streamed in chunks of 128
// (8 S-tiles + 8 Z-tiles of v_wmma_f32_16x16x32_bf16 per iteration).
// Softmax stats reduced with fused VOP2+DPP16 (one instr/stage). P transposed
// through a wave-private LDS tile in K-permuted order (one packed u32 store
// per lane per row); hM carries the matching K permutation.
// ---------------------------------------------------------------------------
__global__ __launch_bounds__(128) void attn_kernel(
    const unsigned short* __restrict__ fT,
    const unsigned short* __restrict__ gT,
    const unsigned short* __restrict__ hM,
    float* __restrict__ zbuf)
{
    __shared__ unsigned int Pl[4 * 16 * 16];     // per-wave 16x32 bf16 P tile

    const int wave = threadIdx.x >> 5;
    const int lane = threadIdx.x & 31;
    const int qb   = blockIdx.x & 63;            // 64 q-blocks per batch
    const int b    = blockIdx.x >> 6;
    const int q0   = qb * 64 + wave * 16;
    const int half = lane >> 4;
    const int l15  = lane & 15;
    const int rowbase = half * 8;
    const int kb  = half * 16;                   // A-frag K interleave
    const int kb2 = half * 16;                   // B-frag K interleave

    // A-fragment of f^T (16 q-rows x K=32), reused across all p.
    BF16Frag aF;
    {
        const unsigned short* base =
            fT + ((size_t)(b * NTOK + q0 + l15)) * MID + kb;
        aF.q[0] = *(const uint4*)(base);
        aF.q[1] = *(const uint4*)(base + 8);
    }

    v8f z0 = {0.f,0.f,0.f,0.f,0.f,0.f,0.f,0.f};
    v8f z1 = z0;
    float mi[8], li[8];
#pragma unroll
    for (int r = 0; r < 8; ++r) { mi[r] = -1e30f; li[r] = 0.f; }

    unsigned int* myP = Pl + wave * (16 * 16);
    const unsigned short* gbase = gT + ((size_t)(b * NTOK + l15)) * MID + kb2;
    const unsigned short* hbase = hM + ((size_t)(b * MID + l15)) * NTOK + kb2;

    for (int p0 = 0; p0 < NTOK; p0 += 128) {
        // --- 8 S-tiles: columns p0 .. p0+127 ---
        const unsigned short* g0 = gbase + (size_t)p0 * MID;
        v8f zero = {0.f,0.f,0.f,0.f,0.f,0.f,0.f,0.f};
        v8f S[8];
#pragma unroll
        for (int j = 0; j < 8; ++j) {
            BF16Frag gB;
            gB.q[0] = *(const uint4*)(g0 + (size_t)j * 16 * MID);
            gB.q[1] = *(const uint4*)(g0 + (size_t)j * 16 * MID + 8);
            S[j] = __builtin_amdgcn_wmma_f32_16x16x32_bf16(
                       false, aF.v, false, gB.v, (short)0, zero, false, false);
        }
        __builtin_prefetch(g0 + (size_t)128 * MID, 0, 3); // next g chunk

        // --- online softmax over this 128-wide p chunk ---
#pragma unroll
        for (int r = 0; r < 8; ++r) {
            float mb = fmaxf(fmaxf(fmaxf(S[0][r], S[1][r]),
                                   fmaxf(S[2][r], S[3][r])),
                             fmaxf(fmaxf(S[4][r], S[5][r]),
                                   fmaxf(S[6][r], S[7][r])));
            mb = redmax16(mb);
            float mnew = fmaxf(mi[r], mb);
            float corr = exp2f((mi[r] - mnew) * LOG2E);
            float srow = 0.f;
#pragma unroll
            for (int j = 0; j < 8; ++j) {
                float e = exp2f((S[j][r] - mnew) * LOG2E);
                S[j][r] = e;                       // reuse S as P
                srow += e;
            }
            srow = redsum16(srow);
            li[r] = li[r] * corr + srow;
            mi[r] = mnew;
            z0[r] *= corr;
            z1[r] *= corr;
        }

        // --- four 32-wide slices: packed-P transpose in LDS, then Z WMMAs ---
#pragma unroll
        for (int jp = 0; jp < 4; ++jp) {
            // logical cols (l15, l15+16) -> adjacent phys slots (2*l15, 2*l15+1)
#pragma unroll
            for (int r = 0; r < 8; ++r)
                myP[(rowbase + r) * 16 + l15] =
                    pack_bf16x2(S[2 * jp][r], S[2 * jp + 1][r]);
            asm volatile("s_wait_dscnt 0" ::: "memory");
            BF16Frag aP;
            {
                const unsigned short* pr =
                    (const unsigned short*)myP + l15 * 32 + kb;
                aP.q[0] = *(const uint4*)(pr);
                aP.q[1] = *(const uint4*)(pr + 8);
            }
            BF16Frag hB0, hB1;
            {
                const unsigned short* h0 = hbase + p0 + jp * 32;
                hB0.q[0] = *(const uint4*)(h0);
                hB0.q[1] = *(const uint4*)(h0 + 8);
                const unsigned short* h1 = h0 + (size_t)16 * NTOK;
                hB1.q[0] = *(const uint4*)(h1);
                hB1.q[1] = *(const uint4*)(h1 + 8);
            }
            z0 = __builtin_amdgcn_wmma_f32_16x16x32_bf16(
                     false, aP.v, false, hB0.v, (short)0, z0, false, false);
            z1 = __builtin_amdgcn_wmma_f32_16x16x32_bf16(
                     false, aP.v, false, hB1.v, (short)0, z1, false, false);
        }
        __builtin_prefetch(hbase + p0 + 128, 0, 3);       // next h chunk
    }

    // --- finalize: z / l, store fp32 [b][q][m] ---
#pragma unroll
    for (int r = 0; r < 8; ++r) {
        float inv = 1.0f / li[r];
        size_t row = (size_t)(b * NTOK + q0 + rowbase + r);
        zbuf[row * MID + l15]      = z0[r] * inv;
        zbuf[row * MID + l15 + 16] = z1[r] * inv;
    }
}

// ---------------------------------------------------------------------------
// Kernel 3: out = v_w @ z^T + v_b + x   (memory-light epilogue)
// ---------------------------------------------------------------------------
__global__ __launch_bounds__(256) void vproj_kernel(
    const float* __restrict__ zbuf, const float* __restrict__ v_w,
    const float* __restrict__ v_b, const float* __restrict__ x,
    float* __restrict__ out)
{
    int t = blockIdx.x * blockDim.x + threadIdx.x;   // one thread per (b,n)
    int b = t >> 12;
    int n = t & (NTOK - 1);
    float zr[MID];
#pragma unroll
    for (int m = 0; m < MID; ++m) zr[m] = zbuf[(size_t)t * MID + m];
    for (int c = 0; c < CIN; ++c) {
        float acc = v_b[c];
#pragma unroll
        for (int m = 0; m < MID; ++m)
            acc = fmaf(v_w[c * MID + m], zr[m], acc);
        size_t idx = ((size_t)(b * CIN + c)) * NTOK + n;
        out[idx] = acc + x[idx];
    }
}

extern "C" void kernel_launch(void* const* d_in, const int* in_sizes, int n_in,
                              void* d_out, int out_size, void* d_ws, size_t ws_size,
                              hipStream_t stream) {
    (void)in_sizes; (void)n_in; (void)out_size; (void)ws_size;
    const float* x       = (const float*)d_in[0];
    const float* f_w     = (const float*)d_in[1];
    const float* f_b     = (const float*)d_in[2];
    const float* f_gamma = (const float*)d_in[3];
    const float* f_beta  = (const float*)d_in[4];
    const float* f_mean  = (const float*)d_in[5];
    const float* f_var   = (const float*)d_in[6];
    const float* g_w     = (const float*)d_in[7];
    const float* g_b     = (const float*)d_in[8];
    const float* g_gamma = (const float*)d_in[9];
    const float* g_beta  = (const float*)d_in[10];
    const float* g_mean  = (const float*)d_in[11];
    const float* g_var   = (const float*)d_in[12];
    const float* h_w     = (const float*)d_in[13];
    const float* h_b     = (const float*)d_in[14];
    const float* v_w     = (const float*)d_in[15];
    const float* v_b     = (const float*)d_in[16];

    char* ws = (char*)d_ws;
    unsigned short* fT = (unsigned short*)(ws);                 // 1 MB
    unsigned short* gT = (unsigned short*)(ws + (1u << 20));    // 1 MB
    unsigned short* hM = (unsigned short*)(ws + (2u << 20));    // 1 MB
    float*          zb = (float*)        (ws + (3u << 20));     // 2 MB

    proj_kernel<<<(4 * NTOK) / 256, 256, 0, stream>>>(
        x, f_w, f_b, f_gamma, f_beta, f_mean, f_var,
        g_w, g_b, g_gamma, g_beta, g_mean, g_var,
        h_w, h_b, fT, gT, hM);

    attn_kernel<<<4 * (NTOK / 64), 128, 0, stream>>>(fT, gT, hM, zb);

    vproj_kernel<<<(4 * NTOK) / 256, 256, 0, stream>>>(zb, v_w, v_b, x,
                                                       (float*)d_out);
}
